// KMultiHeadedAttention_89739046683493
// MI455X (gfx1250) — matrix operand order repaired and checked by reading, not compiled, not run on hardware
//
#include <hip/hip_runtime.h>

typedef __attribute__((ext_vector_type(2)))  float    v2f;
typedef __attribute__((ext_vector_type(8)))  float    v8f;
typedef __attribute__((ext_vector_type(8)))  _Float16 h8;
typedef __attribute__((ext_vector_type(16))) _Float16 v16h;
typedef __attribute__((ext_vector_type(4)))  unsigned int u32x4;
typedef __attribute__((ext_vector_type(4)))  int      i32x4;
typedef __attribute__((ext_vector_type(8)))  int      i32x8;

#define BATCH  4
#define DMODEL 256
#define NHEADS 4
#define HDIM   64
#define SEQ    2048
#define PITCH  72   // halves; 144 B = 9*16 -> b128-aligned rows, bank-spread

static __device__ inline v8f wmma_f32k4(v2f a, v2f b, v8f c) {
  return __builtin_amdgcn_wmma_f32_16x16x4_f32(false, a, false, b, (short)0, c, false, false);
}
static __device__ inline v8f wmma_f16(v16h a, v16h b, v8f c) {
  return __builtin_amdgcn_wmma_f32_16x16x32_f16(false, a, false, b, (short)0, c, false, false);
}
static __device__ inline v16h frag16(const _Float16* lo, const _Float16* hi) {
  h8 a = *(const h8*)lo;
  h8 b = *(const h8*)hi;
  v16h r;
#pragma unroll
  for (int i = 0; i < 8; ++i) { r[i] = a[i]; r[8 + i] = b[i]; }
  return r;
}

// ---------------------------------------------------------------------------
// TDM: DMA a 64x64 2-byte tile (row length 64 elems) into LDS rows of PITCH=72
// halves using the D# pad feature (pad 4 dwords after every 32 dwords stored).
// tensor_d0/stride0 describe the contiguous dimension of the source tensor.
// ---------------------------------------------------------------------------
static __device__ inline void tdm_load_tile64(unsigned int lds_byte,
                                              const void* gptr,
                                              unsigned int tensor_d0,
                                              unsigned int tensor_d1,
                                              unsigned int stride0) {
  unsigned long long ga = (unsigned long long)(size_t)gptr;
  u32x4 g0;
  g0[0] = 1u;                                   // count=1, user D#, no gather
  g0[1] = lds_byte;                             // lds_addr
  g0[2] = (unsigned int)(ga & 0xFFFFFFFFull);   // global_addr[31:0]
  g0[3] = (unsigned int)((ga >> 32) & 0x01FFFFFFull) | (2u << 30);  // type=2
  i32x8 g1;
  g1[0] = (int)((1u << 16)        // data_size = 2 bytes
              | (1u << 20)        // pad_enable
              | (4u << 22)        // pad_interval: 2^(4+1)=32 dwords (64 halves)
              | (3u << 25));      // pad_amount: 4 dwords (8 halves)
  g1[1] = (int)((tensor_d0 & 0xFFFFu) << 16);                  // tensor_dim0 lo
  g1[2] = (int)(((tensor_d0 >> 16) & 0xFFFFu) |
                ((tensor_d1 & 0xFFFFu) << 16));                // d0 hi | d1 lo
  g1[3] = (int)(((tensor_d1 >> 16) & 0xFFFFu) | (64u << 16));  // d1 hi | tile_dim0
  g1[4] = 64;                                                  // tile_dim1=64
  g1[5] = (int)stride0;                                        // dim0 stride lo
  g1[6] = 0;
  g1[7] = 0;
  i32x4 z4 = {0, 0, 0, 0};
  i32x8 z8 = {0, 0, 0, 0, 0, 0, 0, 0};
  __builtin_amdgcn_tensor_load_to_lds(g0, g1, z4, z4, z8, 0);
}

// ---------------------------------------------------------------------------
// Projection: out = W(256x256) @ X(256x2048) + bias via fp32 WMMA 16x16x4.
// MODE 0: Q -> f16 [b,h,n,dd]            (dd contiguous; A-frag friendly)
// MODE 1: K -> f16 [b,h,m,dd], scaled by weight[b,m]/8 (folded into scores)
// MODE 2: V -> f16 [b,h,dd,m]            (natural GEMM layout, coalesced)
// MODE 3: final -> fp32 [b,o,n] (d_out)
// Block: 256 thr = 8 waves; wave computes 16 rows x 64 cols.
// ---------------------------------------------------------------------------
template <int MODE>
__global__ __launch_bounds__(256)
void proj_kernel(const float* __restrict__ W, const float* __restrict__ bias,
                 const float* __restrict__ X, _Float16* __restrict__ outH,
                 float* __restrict__ outF, const float* __restrict__ colScale) {
  const int lane = threadIdx.x & 31;
  const int wave = threadIdx.x >> 5;
  const int r0 = blockIdx.x * 128 + wave * 16;
  const int c0 = blockIdx.y * 64;
  const int b  = blockIdx.z;
  const float* Xb = X + (size_t)b * DMODEL * SEQ;

  const int hl  = (lane < 16) ? 0 : 1;
  const int l15 = lane & 15;
  const int row = r0 + l15;
  const int kb  = hl * 2;

  v8f acc[4];
#pragma unroll
  for (int t = 0; t < 4; ++t)
#pragma unroll
    for (int j = 0; j < 8; ++j) acc[t][j] = 0.0f;

  for (int k = 0; k < DMODEL; k += 4) {
    v2f a = *(const v2f*)&W[row * DMODEL + k + kb];
    const float* x0 = &Xb[(size_t)(k + kb) * SEQ + c0 + l15];
#pragma unroll
    for (int t = 0; t < 4; ++t) {
      v2f bb;
      bb.x = x0[t * 16];
      bb.y = x0[SEQ + t * 16];
      acc[t] = wmma_f32k4(a, bb, acc[t]);
    }
  }

  float bj[8];
#pragma unroll
  for (int j = 0; j < 8; ++j) bj[j] = bias[r0 + j + hl * 8];

#pragma unroll
  for (int t = 0; t < 4; ++t) {
    const int n = c0 + t * 16 + l15;
    float sc = 1.0f;
    if constexpr (MODE == 1) sc = colScale[(size_t)b * SEQ + n] * 0.125f;
#pragma unroll
    for (int j = 0; j < 8; ++j) {
      const int o = r0 + j + hl * 8;
      const float v = acc[t][j] + bj[j];
      if constexpr (MODE == 0) {
        outH[(((size_t)b * NHEADS + (o & 3)) * SEQ + n) * HDIM + (o >> 2)] =
            (_Float16)v;
      } else if constexpr (MODE == 1) {
        outH[(((size_t)b * NHEADS + (o & 3)) * SEQ + n) * HDIM + (o >> 2)] =
            (_Float16)(v * sc);
      } else if constexpr (MODE == 2) {
        outH[(((size_t)b * NHEADS + (o & 3)) * HDIM + (o >> 2)) * SEQ + n] =
            (_Float16)v;
      } else {
        outF[((size_t)b * DMODEL + o) * SEQ + n] = v;
      }
    }
  }
}

// ---------------------------------------------------------------------------
// Flash attention. Q[bh][n][dd] read directly from global into A-fragments.
// K[bh][m][dd] and V[bh][dd][m] tiles DMA'd by the TDM into double-buffered
// LDS (pad feature fills pitch-72 rows); the next tile's DMA overlaps with the
// current tile's WMMA work; one barrier per iteration. Online softmax stats
// live entirely in registers (O orientation M=n,N=dd matches stat slots).
// ---------------------------------------------------------------------------
__global__ __launch_bounds__(256)
void attn_kernel(const _Float16* __restrict__ Qh, const _Float16* __restrict__ Kh,
                 const _Float16* __restrict__ Vh, const float* __restrict__ mask,
                 float* __restrict__ Xo) {
  __shared__ _Float16 sK[2][64 * PITCH];    // [m][dd]
  __shared__ _Float16 sV[2][64 * PITCH];    // [dd][m]
  __shared__ _Float16 sP[8 * 16 * PITCH];   // per-wave [n(16)][m(64)]

  const int tid  = threadIdx.x;
  const int lane = tid & 31;
  const int wave = tid >> 5;
  const int bh   = blockIdx.y;              // b*4 + h
  const int b    = bh >> 2;
  const int h    = bh & 3;
  const int n0   = blockIdx.x * 128;

  const _Float16* Qb = Qh + (size_t)bh * SEQ * HDIM;   // [n][dd]
  const _Float16* Kb = Kh + (size_t)bh * SEQ * HDIM;   // [m][dd]
  const _Float16* Vb = Vh + (size_t)bh * HDIM * SEQ;   // [dd][m]

  // Kick off DMA of tile 0 before doing anything else.
  if (wave == 0) {
    tdm_load_tile64((unsigned int)(size_t)&sK[0][0], Kb, HDIM, SEQ, HDIM);
    tdm_load_tile64((unsigned int)(size_t)&sV[0][0], Vb, SEQ, HDIM, SEQ);
  }

  const int hl  = (lane < 16) ? 0 : 1;
  const int l15 = lane & 15;
  const int nrow = wave * 16 + l15;

  // Q A-fragments straight from global (two b128s each), live all loop.
  v16h qa[2];
#pragma unroll
  for (int kk = 0; kk < 2; ++kk) {
    const _Float16* qp = &Qb[(size_t)(n0 + nrow) * HDIM + kk * 32 + hl * 8];
    qa[kk] = frag16(qp, qp + 16);
  }

  float rM[8], rL[8];
#pragma unroll
  for (int j = 0; j < 8; ++j) { rM[j] = -1e30f; rL[j] = 0.0f; }
  v8f oacc[4];
#pragma unroll
  for (int t = 0; t < 4; ++t)
#pragma unroll
    for (int j = 0; j < 8; ++j) oacc[t][j] = 0.0f;

  _Float16* myP = &sP[wave * 16 * PITCH];

  for (int it = 0; it < SEQ / 64; ++it) {
    const int mt  = it * 64;
    const int buf = it & 1;

    if (wave == 0) __builtin_amdgcn_s_wait_tensorcnt(0);
    __syncthreads();  // tile `it` visible; everyone done with buffer buf^1

    // Overlap: DMA tile it+1 into the other buffer while we compute.
    if (wave == 0 && mt + 64 < SEQ) {
      tdm_load_tile64((unsigned int)(size_t)&sK[buf ^ 1][0],
                      Kb + (size_t)(mt + 64) * HDIM, HDIM, SEQ, HDIM);
      tdm_load_tile64((unsigned int)(size_t)&sV[buf ^ 1][0],
                      Vb + (mt + 64), SEQ, HDIM, SEQ);
    }

    const _Float16* kt = sK[buf];
    const _Float16* vt = sV[buf];

    // S(16n x 64m) = Q * K^T ; K already carries weight[m]/sqrt(d).
    v8f s[4];
#pragma unroll
    for (int t = 0; t < 4; ++t) {
      v8f a;
#pragma unroll
      for (int j = 0; j < 8; ++j) a[j] = 0.0f;
#pragma unroll
      for (int kk = 0; kk < 2; ++kk) {
        const _Float16* base = &kt[(t * 16 + l15) * PITCH + kk * 32 + hl * 8];
        v16h kf = frag16(base, base + 16);
        a = wmma_f16(qa[kk], kf, a);
      }
      s[t] = a;
    }

    // Multiplicative mask + per-row tile max.
    float tmax[8];
#pragma unroll
    for (int j = 0; j < 8; ++j) tmax[j] = -1e30f;
    const int nglob = n0 + wave * 16 + hl * 8;
#pragma unroll
    for (int t = 0; t < 4; ++t) {
      const int m = mt + t * 16 + l15;
#pragma unroll
      for (int j = 0; j < 8; ++j) {
        float mk = mask[((size_t)b * SEQ + (nglob + j)) * SEQ + m];
        float v = s[t][j] * mk;
        s[t][j] = v;
        tmax[j] = fmaxf(tmax[j], v);
      }
    }
#pragma unroll
    for (int d = 1; d < 16; d <<= 1)
#pragma unroll
      for (int j = 0; j < 8; ++j)
        tmax[j] = fmaxf(tmax[j], __shfl_xor(tmax[j], d, 32));

    // Online softmax; O rescale is a pure per-VGPR multiply (matched slots).
    float alpha[8];
#pragma unroll
    for (int j = 0; j < 8; ++j) {
      float nm = fmaxf(rM[j], tmax[j]);
      alpha[j] = __expf(rM[j] - nm);
      rM[j] = nm;
    }
#pragma unroll
    for (int t = 0; t < 4; ++t)
#pragma unroll
      for (int j = 0; j < 8; ++j) oacc[t][j] *= alpha[j];

    float psum[8];
#pragma unroll
    for (int j = 0; j < 8; ++j) psum[j] = 0.0f;
#pragma unroll
    for (int t = 0; t < 4; ++t)
#pragma unroll
      for (int j = 0; j < 8; ++j) {
        float p = __expf(s[t][j] - rM[j]);
        psum[j] += p;
        myP[(j + hl * 8) * PITCH + t * 16 + l15] = (_Float16)p;  // C/D -> [n][m]
      }
#pragma unroll
    for (int d = 1; d < 16; d <<= 1)
#pragma unroll
      for (int j = 0; j < 8; ++j) psum[j] += __shfl_xor(psum[j], d, 32);
#pragma unroll
    for (int j = 0; j < 8; ++j) rL[j] = rL[j] * alpha[j] + psum[j];

    __builtin_amdgcn_wave_barrier();  // wave-private P staging; DS is in-order

    // O(16n x 64dd) += P * V
#pragma unroll
    for (int t = 0; t < 4; ++t) {
#pragma unroll
      for (int kk = 0; kk < 2; ++kk) {
        const _Float16* pb = &myP[l15 * PITCH + kk * 32 + hl * 8];
        v16h pA = frag16(pb, pb + 16);
        const _Float16* vb2 = &vt[(t * 16 + l15) * PITCH + kk * 32 + hl * 8];
        v16h vB = frag16(vb2, vb2 + 16);
        oacc[t] = wmma_f16(pA, vB, oacc[t]);
      }
    }
  }

  // Normalize and store x[b, dd*4+h, n] fp32 for the final projection.
#pragma unroll
  for (int j = 0; j < 8; ++j) {
    const float inv = 1.0f / rL[j];
    const int n = n0 + wave * 16 + j + hl * 8;
#pragma unroll
    for (int t = 0; t < 4; ++t) {
      const int dd = t * 16 + l15;
      Xo[((size_t)b * DMODEL + (dd * NHEADS + h)) * SEQ + n] = oacc[t][j] * inv;
    }
  }
}

// ---------------------------------------------------------------------------
extern "C" void kernel_launch(void* const* d_in, const int* in_sizes, int n_in,
                              void* d_out, int out_size, void* d_ws, size_t ws_size,
                              hipStream_t stream) {
  const float* query  = (const float*)d_in[0];
  const float* key    = (const float*)d_in[1];
  const float* value  = (const float*)d_in[2];
  const float* weight = (const float*)d_in[3];
  const float* mask   = (const float*)d_in[4];
  const float* Wq = (const float*)d_in[5];
  const float* bq = (const float*)d_in[6];
  const float* Wk = (const float*)d_in[7];
  const float* bk = (const float*)d_in[8];
  const float* Wv = (const float*)d_in[9];
  const float* bv = (const float*)d_in[10];
  const float* Wm = (const float*)d_in[11];
  const float* bm = (const float*)d_in[12];
  float* out = (float*)d_out;

  const size_t headElems = (size_t)BATCH * NHEADS * HDIM * SEQ;  // 2M halves
  _Float16* Qh = (_Float16*)d_ws;
  _Float16* Kh = Qh + headElems;
  _Float16* Vh = Kh + headElems;
  float*    Xw = (float*)(Vh + headElems);  // 8 MB fp32; 20 MB ws total

  dim3 pg(2, 32, BATCH), pb(256);
  proj_kernel<0><<<pg, pb, 0, stream>>>(Wq, bq, query, Qh, nullptr, nullptr);
  proj_kernel<1><<<pg, pb, 0, stream>>>(Wk, bk, key,   Kh, nullptr, weight);
  proj_kernel<2><<<pg, pb, 0, stream>>>(Wv, bv, value, Vh, nullptr, nullptr);

  dim3 ag(SEQ / 128, BATCH * NHEADS), ab(256);
  attn_kernel<<<ag, ab, 0, stream>>>(Qh, Kh, Vh, mask, Xw);

  proj_kernel<3><<<pg, pb, 0, stream>>>(Wm, bm, Xw, nullptr, out, nullptr);
}
